// AdaConvNeXt_10471130268344
// MI455X (gfx1250) — matrix operand order repaired and checked by reading, not compiled, not run on hardware
//
#include <hip/hip_runtime.h>
#include <hip/hip_bf16.h>
#include <math.h>

typedef _Float16 v8h  __attribute__((ext_vector_type(8)));
typedef _Float16 v16h __attribute__((ext_vector_type(16)));
typedef float    v8f  __attribute__((ext_vector_type(8)));
typedef unsigned int u32x4 __attribute__((ext_vector_type(4)));
typedef int          i32x4 __attribute__((ext_vector_type(4)));
typedef int          i32x8 __attribute__((ext_vector_type(8)));

#define BB 128
#define CCH 384
#define HH 28
#define WW 28
#define NN 784
#define NKEEP 392
#define C4 1536
#define LN_EPS 1e-6f

#define CPB 8          // channels per conv block
#define PLANE 1157     // 34*34 padded (bank-conflict-free channel stride)
#define AS_STRIDE 392  // 384 + 8 f16 pad -> 196 dwords (= 4 mod 64), 16B aligned rows
#define HS_STRIDE 1544 // 1536 + 8 f16 pad -> 772 dwords (= 4 mod 64)

#if defined(__HIP_DEVICE_COMPILE__) && defined(__has_builtin)
#if __has_builtin(__builtin_amdgcn_tensor_load_to_lds) && __has_builtin(__builtin_amdgcn_s_wait_tensorcnt)
#define HAVE_TDM 1
#endif
#endif
#if __has_include(<hip/amd_detail/amd_gfx1250_TDM.h>)
#define TDM_ARITY6 1
#endif

static __device__ __forceinline__ v16h mk16(v8h lo, v8h hi8) {
  v16h r;
#pragma unroll
  for (int i = 0; i < 8; ++i) { r[i] = lo[i]; r[i + 8] = hi8[i]; }
  return r;
}

#define WMMA_F16(A, Bv, Cv) \
  __builtin_amdgcn_wmma_f32_16x16x32_f16(false, (A), false, (Bv), (short)0, (Cv), false, false)

// ---------------------------------------------------------------------------
// Pack an fp32 row-major (K x Nw) weight into f16 WMMA B-fragment layout:
// fragment(nt,kt) = 32 lanes x 16 f16, lane = (N-col in tile) + 16*(K-half),
// element j -> K = kt*32 + (j&8)*2 + (lane>>4)*8 + (j&7), N = nt*16 + (lane&15).
// ---------------------------------------------------------------------------
__global__ void pack_w_f16(const float* __restrict__ W, _Float16* __restrict__ out,
                           int K, int Nw) {
  int KT = K >> 5;
  long total = (long)(Nw >> 4) * KT * 512;
  for (long e = (long)blockIdx.x * blockDim.x + threadIdx.x; e < total;
       e += (long)gridDim.x * blockDim.x) {
    int j    = (int)(e & 15);
    int lane = (int)((e >> 4) & 31);
    long blk = e >> 9;
    int kt = (int)(blk % KT);
    int nt = (int)(blk / KT);
    int k = kt * 32 + (j & 8) * 2 + (lane >> 4) * 8 + (j & 7);
    int n = nt * 16 + (lane & 15);
    out[e] = (_Float16)W[(long)k * Nw + n];
  }
}

// ---------------------------------------------------------------------------
// Depthwise 7x7 conv (pad 3) + bias, NCHW in -> NHWC out (t[b][n][c]).
// ---------------------------------------------------------------------------
__global__ void __launch_bounds__(256) dwconv_kernel(const float* __restrict__ x,
    const float* __restrict__ dw_w, const float* __restrict__ dw_b,
    float* __restrict__ t) {
  __shared__ float tile[CPB * PLANE];
  int bx = blockIdx.x;
  int b  = bx / (CCH / CPB);
  int cb = (bx % (CCH / CPB)) * CPB;

  for (int it = threadIdx.x; it < CPB * 34 * 34; it += 256) {
    int c = it / (34 * 34);
    int p = it % (34 * 34);
    int ih = p / 34 - 3, iw = p % 34 - 3;
    float v = 0.f;
    if (ih >= 0 && ih < HH && iw >= 0 && iw < WW)
      v = x[(((long)b * CCH + cb + c) * HH + ih) * WW + iw];
    tile[c * PLANE + p] = v;
  }
  __syncthreads();

  int c  = threadIdx.x & (CPB - 1);
  int ty = threadIdx.x >> 3;
  float wreg[49];
#pragma unroll
  for (int i = 0; i < 49; ++i) wreg[i] = dw_w[(cb + c) * 49 + i];
  float bias = dw_b[cb + c];

  for (int p = ty; p < NN; p += 32) {
    int h = p / WW, w = p % WW;
    const float* tp = &tile[c * PLANE + h * 34 + w];
    float acc = bias;
#pragma unroll
    for (int kh = 0; kh < 7; ++kh)
#pragma unroll
      for (int kw = 0; kw < 7; ++kw)
        acc += tp[kh * 34 + kw] * wreg[kh * 7 + kw];
    t[((long)b * NN + p) * CCH + cb + c] = acc;
  }
}

// ---------------------------------------------------------------------------
// Shared LN prologue: gather 16 token rows, LayerNorm, write f16 panel to LDS.
// ---------------------------------------------------------------------------
static __device__ __forceinline__ void ln_prologue(
    const float* __restrict__ t, const int* __restrict__ idx,
    const float* __restrict__ lw, const float* __restrict__ lb,
    _Float16* As, float (*red_s)[16], float (*red_q)[16],
    float* mu_s, float* rs_s, int* tokbuf, int* bbuf) {
  int tid = threadIdx.x;
  int row = tid >> 4, lanei = tid & 15;
  long g = (long)blockIdx.x * 16 + row;
  int b = (int)(g / NKEEP);
  int j = (int)(g % NKEEP);
  int token = idx[b * NKEEP + j];
  if (lanei == 0) { tokbuf[row] = token; bbuf[row] = b; }
  const float* trow = t + ((long)b * NN + token) * CCH;

  float s = 0.f, q = 0.f;
  for (int k = lanei; k < CCH; k += 16) { float v = trow[k]; s += v; q += v * v; }
  red_s[row][lanei] = s; red_q[row][lanei] = q;
  __syncthreads();
  if (lanei == 0) {
    float ss = 0.f, qq = 0.f;
#pragma unroll
    for (int i = 0; i < 16; ++i) { ss += red_s[row][i]; qq += red_q[row][i]; }
    float mu = ss / CCH;
    float var = qq / CCH - mu * mu;
    mu_s[row] = mu; rs_s[row] = rsqrtf(var + LN_EPS);
  }
  __syncthreads();
  float mu = mu_s[row], rs = rs_s[row];
  for (int k = lanei; k < CCH; k += 16) {
    float v = (trow[k] - mu) * rs * lw[k] + lb[k];
    As[row * AS_STRIDE + k] = (_Float16)v;
  }
  __syncthreads();
}

// ---------------------------------------------------------------------------
// Branch 1: LN -> GEMM 384->1536 (+bias, exact GELU, h kept in LDS as f16)
//            -> GEMM 1536->384 (+bias, *gamma) -> scatter to r[b][tok][c].
// Dual/triple accumulator chains per wave for WMMA ILP; A fragments shared.
// ---------------------------------------------------------------------------
__global__ void __launch_bounds__(256) mlp_kernel(
    const float* __restrict__ t, const int* __restrict__ idx1,
    const float* __restrict__ norm_w, const float* __restrict__ norm_b,
    const _Float16* __restrict__ w1p, const float* __restrict__ b1,
    const _Float16* __restrict__ w2p, const float* __restrict__ b2,
    const float* __restrict__ gamma, float* __restrict__ r) {
  __shared__ _Float16 As[16 * AS_STRIDE];   // 12544 B
  __shared__ _Float16 Hs[16 * HS_STRIDE];   // 49408 B
  __shared__ float red_s[16][16];
  __shared__ float red_q[16][16];
  __shared__ float mu_s[16], rs_s[16];
  __shared__ int tokbuf[16], bbuf[16];

  ln_prologue(t, idx1, norm_w, norm_b, As, red_s, red_q, mu_s, rs_s, tokbuf, bbuf);

  int tid = threadIdx.x;
  int wave = tid >> 5, lane = tid & 31;
  int m16 = lane & 15, hi = lane >> 4;

  // ---- GEMM1: [16x384]x[384x1536]; 2 N-tiles per wave concurrently ----
  for (int p = 0; p < 6; ++p) {
    int nt0 = wave + 16 * p;                // nt1 = nt0 + 8
    v8f acc0 = {}, acc1 = {};
    const _Float16* bb0 = w1p + ((long)nt0 * 12) * 512;
    const _Float16* bb1 = bb0 + (long)8 * 12 * 512;
#pragma unroll 3
    for (int kt = 0; kt < 12; ++kt) {
      const _Float16* ap = &As[m16 * AS_STRIDE + kt * 32 + hi * 8];
      v16h a = mk16(*(const v8h*)(ap), *(const v8h*)(ap + 16));
      const _Float16* bp0 = bb0 + (long)kt * 512 + lane * 16;
      const _Float16* bp1 = bb1 + (long)kt * 512 + lane * 16;
      if (kt + 1 < 12) {
        __builtin_prefetch((const void*)(bp0 + 512), 0, 0);
        __builtin_prefetch((const void*)(bp1 + 512), 0, 0);
      }
      v16h bf0 = mk16(*(const v8h*)(bp0), *(const v8h*)(bp0 + 8));
      v16h bf1 = mk16(*(const v8h*)(bp1), *(const v8h*)(bp1 + 8));
      acc0 = WMMA_F16(a, bf0, acc0);
      acc1 = WMMA_F16(a, bf1, acc1);
    }
#pragma unroll
    for (int v = 0; v < 8; ++v) {
      int mrow = v + 8 * hi;
      int n0 = nt0 * 16 + m16, n1 = n0 + 128;
      float x0 = acc0[v] + b1[n0];
      float x1 = acc1[v] + b1[n1];
      float g0 = 0.5f * x0 * (1.f + erff(x0 * 0.70710678118f));
      float g1 = 0.5f * x1 * (1.f + erff(x1 * 0.70710678118f));
      Hs[mrow * HS_STRIDE + n0] = (_Float16)g0;
      Hs[mrow * HS_STRIDE + n1] = (_Float16)g1;
    }
  }
  __syncthreads();

  // ---- GEMM2: [16x1536]x[1536x384]; all 3 N-tiles of this wave at once ----
  {
    int nt0 = wave, nt1 = wave + 8, nt2 = wave + 16;
    v8f acc0 = {}, acc1 = {}, acc2 = {};
    const _Float16* bb0 = w2p + ((long)nt0 * 48) * 512;
    const _Float16* bb1 = w2p + ((long)nt1 * 48) * 512;
    const _Float16* bb2 = w2p + ((long)nt2 * 48) * 512;
#pragma unroll 2
    for (int kt = 0; kt < 48; ++kt) {
      const _Float16* ap = &Hs[m16 * HS_STRIDE + kt * 32 + hi * 8];
      v16h a = mk16(*(const v8h*)(ap), *(const v8h*)(ap + 16));
      const _Float16* bp0 = bb0 + (long)kt * 512 + lane * 16;
      const _Float16* bp1 = bb1 + (long)kt * 512 + lane * 16;
      const _Float16* bp2 = bb2 + (long)kt * 512 + lane * 16;
      if (kt + 1 < 48) __builtin_prefetch((const void*)(bp0 + 512), 0, 0);
      v16h bf0 = mk16(*(const v8h*)(bp0), *(const v8h*)(bp0 + 8));
      v16h bf1 = mk16(*(const v8h*)(bp1), *(const v8h*)(bp1 + 8));
      v16h bf2 = mk16(*(const v8h*)(bp2), *(const v8h*)(bp2 + 8));
      acc0 = WMMA_F16(a, bf0, acc0);
      acc1 = WMMA_F16(a, bf1, acc1);
      acc2 = WMMA_F16(a, bf2, acc2);
    }
#pragma unroll
    for (int v = 0; v < 8; ++v) {
      int mrow = v + 8 * hi;
      long rbase = ((long)bbuf[mrow] * NN + tokbuf[mrow]) * CCH;
      int n0 = nt0 * 16 + m16, n1 = nt1 * 16 + m16, n2 = nt2 * 16 + m16;
      __builtin_nontemporal_store((acc0[v] + b2[n0]) * gamma[n0], &r[rbase + n0]);
      __builtin_nontemporal_store((acc1[v] + b2[n1]) * gamma[n1], &r[rbase + n1]);
      __builtin_nontemporal_store((acc2[v] + b2[n2]) * gamma[n2], &r[rbase + n2]);
    }
  }
}

// ---------------------------------------------------------------------------
// Branch 2: LN -> [16x384]x[384x384] GEMM (+fp_b, *fp_gamma) -> scatter.
// ---------------------------------------------------------------------------
__global__ void __launch_bounds__(256) fp_kernel(
    const float* __restrict__ t, const int* __restrict__ idx2,
    const float* __restrict__ fnw, const float* __restrict__ fnb,
    const _Float16* __restrict__ fwp, const float* __restrict__ fpb,
    const float* __restrict__ fpg, float* __restrict__ r) {
  __shared__ _Float16 As[16 * AS_STRIDE];
  __shared__ float red_s[16][16];
  __shared__ float red_q[16][16];
  __shared__ float mu_s[16], rs_s[16];
  __shared__ int tokbuf[16], bbuf[16];

  ln_prologue(t, idx2, fnw, fnb, As, red_s, red_q, mu_s, rs_s, tokbuf, bbuf);

  int tid = threadIdx.x;
  int wave = tid >> 5, lane = tid & 31;
  int m16 = lane & 15, hi = lane >> 4;

  int nt0 = wave, nt1 = wave + 8, nt2 = wave + 16;
  v8f acc0 = {}, acc1 = {}, acc2 = {};
  const _Float16* bb0 = fwp + ((long)nt0 * 12) * 512;
  const _Float16* bb1 = fwp + ((long)nt1 * 12) * 512;
  const _Float16* bb2 = fwp + ((long)nt2 * 12) * 512;
#pragma unroll 3
  for (int kt = 0; kt < 12; ++kt) {
    const _Float16* ap = &As[m16 * AS_STRIDE + kt * 32 + hi * 8];
    v16h a = mk16(*(const v8h*)(ap), *(const v8h*)(ap + 16));
    const _Float16* bp0 = bb0 + (long)kt * 512 + lane * 16;
    const _Float16* bp1 = bb1 + (long)kt * 512 + lane * 16;
    const _Float16* bp2 = bb2 + (long)kt * 512 + lane * 16;
    if (kt + 1 < 12) __builtin_prefetch((const void*)(bp0 + 512), 0, 0);
    v16h bf0 = mk16(*(const v8h*)(bp0), *(const v8h*)(bp0 + 8));
    v16h bf1 = mk16(*(const v8h*)(bp1), *(const v8h*)(bp1 + 8));
    v16h bf2 = mk16(*(const v8h*)(bp2), *(const v8h*)(bp2 + 8));
    acc0 = WMMA_F16(a, bf0, acc0);
    acc1 = WMMA_F16(a, bf1, acc1);
    acc2 = WMMA_F16(a, bf2, acc2);
  }
#pragma unroll
  for (int v = 0; v < 8; ++v) {
    int mrow = v + 8 * hi;
    long rbase = ((long)bbuf[mrow] * NN + tokbuf[mrow]) * CCH;
    int n0 = nt0 * 16 + m16, n1 = nt1 * 16 + m16, n2 = nt2 * 16 + m16;
    __builtin_nontemporal_store((acc0[v] + fpb[n0]) * fpg[n0], &r[rbase + n0]);
    __builtin_nontemporal_store((acc1[v] + fpb[n1]) * fpg[n1], &r[rbase + n1]);
    __builtin_nontemporal_store((acc2[v] + fpb[n2]) * fpg[n2], &r[rbase + n2]);
  }
}

// ---------------------------------------------------------------------------
// out[b][c][n] = x[b][c][n] + r[b][n][c].
// The 16x64 fp32 r-tile (row stride 384 dwords) is fetched by the Tensor Data
// Mover into LDS, using TDM padding (interval 64 dwords, pad 1 dword) to make
// the LDS row stride 65 dwords (bank-conflict-free), then transposed + added.
// ---------------------------------------------------------------------------
__global__ void __launch_bounds__(256) fuse_out_kernel(const float* __restrict__ x,
    const float* __restrict__ r, float* __restrict__ out) {
  __shared__ float tl[16][65];
  int bx = blockIdx.x;
  int b = bx / (6 * 49);
  int rem = bx % (6 * 49);
  int cb = (rem / 49) * 64;
  int n0 = (rem % 49) * 16;
  const float* gsrc = r + ((long)b * NN + n0) * CCH + cb;

#if defined(HAVE_TDM)
  if (threadIdx.x < 32) {
    unsigned lds_off = (unsigned)(unsigned long long)(__SIZE_TYPE__)&tl[0][0];
    unsigned long long ga = (unsigned long long)(__SIZE_TYPE__)gsrc;
    // D# group 0: count=1, lds_addr, global_addr[56:0], type=2
    u32x4 g0;
    g0[0] = 1u;
    g0[1] = lds_off;
    g0[2] = (unsigned)ga;
    g0[3] = (unsigned)((ga >> 32) & 0x1FFFFFFull) | (2u << 30);
    // D# group 1: data_size=4B(2), pad_enable, pad_interval=5 (64 dwords),
    // pad_amount=0 (1 dword); tensor_dim0=384, tensor_dim1=16,
    // tile_dim0=64, tile_dim1=16; tensor_dim0_stride=384.
    i32x8 g1;
    g1[0] = (2 << 16) | (1 << 20) | (5 << 22);
    g1[1] = (384 << 16);       // [31:16] = tensor_dim0[15:0]
    g1[2] = (16 << 16);        // [15:0]=dim0[31:16]=0, [31:16]=tensor_dim1[15:0]
    g1[3] = (64 << 16);        // [15:0]=dim1[31:16]=0, [31:16]=tile_dim0
    g1[4] = 16;                // [15:0]=tile_dim1, [31:16]=tile_dim2=0
    g1[5] = 384;               // tensor_dim0_stride[31:0]
    g1[6] = 0;
    g1[7] = 0;
    i32x4 gz = {0, 0, 0, 0};   // groups 2/3 unused (2D tile)
#if defined(TDM_ARITY6)
    i32x8 gz8 = {0, 0, 0, 0, 0, 0, 0, 0};
    __builtin_amdgcn_tensor_load_to_lds(g0, g1, gz, gz, gz8, 0);
#else
    __builtin_amdgcn_tensor_load_to_lds(g0, g1, gz, gz, 0);
#endif
    __builtin_amdgcn_s_wait_tensorcnt(0);
  }
  __syncthreads();
#else
  for (int it = threadIdx.x; it < 16 * 64; it += 256) {
    int i = it >> 6, jc = it & 63;
    tl[i][jc] = gsrc[(long)i * CCH + jc];
  }
  __syncthreads();
#endif

  for (int it = threadIdx.x; it < 64 * 16; it += 256) {
    int cc = it >> 4, jj = it & 15;
    long oidx = ((long)(b * CCH + cb + cc)) * NN + n0 + jj;
    __builtin_nontemporal_store(x[oidx] + tl[jj][cc], &out[oidx]);
  }
}

// ---------------------------------------------------------------------------
extern "C" void kernel_launch(void* const* d_in, const int* in_sizes, int n_in,
                              void* d_out, int out_size, void* d_ws, size_t ws_size,
                              hipStream_t stream) {
  const float* x      = (const float*)d_in[0];
  const int*   idx1   = (const int*)d_in[1];
  const int*   idx2   = (const int*)d_in[2];
  const float* dw_w   = (const float*)d_in[3];
  const float* dw_b   = (const float*)d_in[4];
  const float* norm_w = (const float*)d_in[5];
  const float* norm_b = (const float*)d_in[6];
  const float* w1     = (const float*)d_in[7];
  const float* b1     = (const float*)d_in[8];
  const float* w2     = (const float*)d_in[9];
  const float* b2     = (const float*)d_in[10];
  const float* gamma  = (const float*)d_in[11];
  const float* fp_nw  = (const float*)d_in[12];
  const float* fp_nb  = (const float*)d_in[13];
  const float* fp_w   = (const float*)d_in[14];
  const float* fp_b   = (const float*)d_in[15];
  const float* fp_g   = (const float*)d_in[16];
  float* out = (float*)d_out;

  char* ws = (char*)d_ws;
  size_t off = 0;
  float* t = (float*)(ws + off);         off += (size_t)BB * NN * CCH * 4;  // conv out NHWC
  float* r = (float*)(ws + off);         off += (size_t)BB * NN * CCH * 4;  // branch results NHWC
  _Float16* w1p = (_Float16*)(ws + off); off += (size_t)CCH * C4 * 2;
  _Float16* w2p = (_Float16*)(ws + off); off += (size_t)C4 * CCH * 2;
  _Float16* fwp = (_Float16*)(ws + off); off += (size_t)CCH * CCH * 2;

  pack_w_f16<<<2304, 256, 0, stream>>>(w1, w1p, CCH, C4);
  pack_w_f16<<<2304, 256, 0, stream>>>(w2, w2p, C4, CCH);
  pack_w_f16<<<576,  256, 0, stream>>>(fp_w, fwp, CCH, CCH);

  dwconv_kernel<<<BB * (CCH / CPB), 256, 0, stream>>>(x, dw_w, dw_b, t);

  mlp_kernel<<<(BB * NKEEP) / 16, 256, 0, stream>>>(t, idx1, norm_w, norm_b,
                                                    w1p, b1, w2p, b2, gamma, r);
  fp_kernel<<<(BB * NKEEP) / 16, 256, 0, stream>>>(t, idx2, fp_nw, fp_nb,
                                                   fwp, fp_b, fp_g, r);

  fuse_out_kernel<<<BB * 6 * 49, 256, 0, stream>>>(x, r, out);
}
